// RetinaNet_50448685859478
// MI455X (gfx1250) — matrix operand order repaired and checked by reading, not compiled, not run on hardware
//
#include <hip/hip_runtime.h>
#include <hip/hip_bf16.h>
#include <cstdint>
#include <cstddef>

// ---------------------------------------------------------------------------
// Types for CDNA5 WMMA
// ---------------------------------------------------------------------------
typedef __attribute__((ext_vector_type(16))) __bf16 v16bf;
typedef __attribute__((ext_vector_type(8)))  __bf16 v8bf;
typedef __attribute__((ext_vector_type(8)))  float  v8f;

#define RN_N        2
#define RN_C        256
#define RN_A        9
#define RN_K        80
#define RN_SCORE_CH (RN_A * RN_K)   // 720
#define RN_PRED_CH  (RN_A * 4)      // 36
#define RN_KTOT     (RN_C * 9)      // 2304
#define RN_TOPK     1000
#define RN_CAP      2048            // per (img, level) candidate cap
#define RN_MAXDET   100
#define RN_IMG_W    1216.0f
#define RN_IMG_H    800.0f
#define RN_SCORE_TH 0.05f
#define RN_NMS_TH   0.5f
#define RN_SCALE_CLAMP 4.135166556742356f  // log(1000/16)

#define RN_KSLAB    128             // K staged per barrier round (divides 256)
#define RN_APAD     136             // padded LDS row: 272B = 17*16B, bank-friendly

// ---------------------------------------------------------------------------
// NCHW f32 -> NHWC bf16 feature convert (channels innermost for the conv)
// ---------------------------------------------------------------------------
__global__ void rn_f2bf_nhwc(const float* __restrict__ src, __bf16* __restrict__ dst,
                             int HW) {
    const int total = RN_N * HW * RN_C;
    for (int i = blockIdx.x * blockDim.x + threadIdx.x; i < total; i += gridDim.x * blockDim.x) {
        int c = i & (RN_C - 1);
        int t2 = i >> 8;
        int n = t2 / HW;
        int p = t2 - n * HW;
        dst[i] = (__bf16)src[((size_t)n * RN_C + c) * HW + p];
    }
}

// ---------------------------------------------------------------------------
// Weight repack: [Cout][256][3][3] f32 -> bf16 WtT[CoutPad][Ktot]
//   row = cout (zero-padded to CoutPad), k = tap*256 + ci (tap-major).
// Each lane's WMMA B-fragment becomes two 16-byte runs of this buffer,
// loadable directly from global with no LDS transpose and no bounds checks.
// ---------------------------------------------------------------------------
__global__ void rn_wtransform(const float* __restrict__ Wsrc, __bf16* __restrict__ WtT,
                              int Cout, int CoutPad) {
    const int tot = CoutPad * RN_KTOT;
    for (int i = blockIdx.x * blockDim.x + threadIdx.x; i < tot; i += gridDim.x * blockDim.x) {
        int co = i / RN_KTOT;
        int r = i - co * RN_KTOT;
        int tap = r >> 8;
        int ci = r & 255;
        __bf16 v = (__bf16)0.0f;
        if (co < Cout) v = (__bf16)Wsrc[((size_t)co * RN_C + ci) * 9 + tap];
        WtT[i] = v;
    }
}

// ---------------------------------------------------------------------------
// Implicit-GEMM 3x3 SAME conv via V_WMMA_F32_16X16X32_BF16 (NHWC).
//   M = Nb*H*W, N = Cout, K = 9*256 tap-major.
//   WG = 256 threads = 8 waves; tile 64(M) x 128(N).
//   128-wide K slab staged per barrier round (16 WMMAs per barrier pair);
//   B fragments loaded directly from global (two 16B runs per lane, k offsets 8h and 16+8h);
//   LDS rows padded to 272B so fragment reads hit 16 distinct banks.
// ---------------------------------------------------------------------------
__global__ __launch_bounds__(256, 2)
void rn_conv3x3_wmma(const __bf16* __restrict__ X,    // [Nb,H,W,256] bf16
                     const __bf16* __restrict__ WtT,  // [CoutPad][2304] bf16
                     const float*  __restrict__ bias, // [Cout]
                     __bf16* __restrict__ Ybf,        // NHWC bf16 out (or null)
                     float*  __restrict__ Yf,         // NHWC f32 out (or null)
                     int Nb, int H, int Wd, int Cout, int relu) {
    const int Cin  = RN_C;
    const int HW   = H * Wd;
    const int Mtot = Nb * HW;
    const int Ktot = RN_KTOT;
    const int mBase = blockIdx.x * 64;
    const int nBase = blockIdx.y * 128;

    __shared__ __align__(16) __bf16 As[64][RN_APAD];  // im2col slab [m][k0..127]

    const int t      = threadIdx.x;
    const int lane   = t & 31;
    const int wv     = t >> 5;
    const int waveM  = wv & 3;
    const int waveN  = wv >> 2;
    const int laneHi = lane >> 4;
    const int l16    = lane & 15;

    // ---- A staging: thread -> (row, 4 x 8-element chunks spanning 128 k)
    const int aRow   = t >> 2;        // 0..63
    const int aChunk = (t & 3) * 8;   // 0,8,16,24 (then +32*j)
    const int mA = mBase + aRow;
    const bool mOK = (mA < Mtot);
    int nImg = 0, yA = 0, xA = 0;
    if (mOK) {
        nImg = mA / HW;
        int rem = mA - nImg * HW;
        yA = rem / Wd;
        xA = rem - yA * Wd;
    }
    const long imgBase = (long)nImg * HW * Cin;

    // ---- B fragment base pointers: two 8-bf16 runs per lane per subtile (k +8h, +16+8h)
    const __bf16* bPtr[4];
#pragma unroll
    for (int s = 0; s < 4; ++s) {
        const int col = nBase + waveN * 64 + s * 16 + l16;
        bPtr[s] = WtT + (long)col * Ktot + laneHi * 8;
    }

    const int rowA = waveM * 16 + l16;
    v8f acc[4] = {{}, {}, {}, {}};

    for (int kBase = 0; kBase < Ktot; kBase += RN_KSLAB) {
        const int tap    = kBase >> 8;   // 128-slabs never straddle a tap
        const int ciBase = kBase & 255;
        const int dy = tap / 3 - 1;      // wave-uniform scalars
        const int dx = tap - (tap / 3) * 3 - 1;

        __syncthreads();
        // ---- stage A: four guarded 16B global loads + 16B LDS stores
        {
            const int yy = yA + dy, xx = xA + dx;
            const bool inb = mOK && (unsigned)yy < (unsigned)H && (unsigned)xx < (unsigned)Wd;
            const __bf16* src = X + imgBase + ((long)yy * Wd + xx) * Cin + ciBase + aChunk;
#pragma unroll
            for (int j = 0; j < 4; ++j) {
                v8bf v = {};
                if (inb) v = *(const v8bf*)(src + 32 * j);
                *(v8bf*)&As[aRow][aChunk + 32 * j] = v;
            }
        }
        __syncthreads();

        // ---- 4 sub-steps of K=32: 16 WMMAs per barrier round
#pragma unroll
        for (int sub = 0; sub < 4; ++sub) {
            const int kk = kBase + sub * 32;
            union { v16bf v; v8bf h[2]; } a, b0, b1, b2, b3;
            a.h[0] = *(const v8bf*)&As[rowA][sub * 32 + laneHi * 8];       // K {0..7}/{8..15}
            a.h[1] = *(const v8bf*)&As[rowA][sub * 32 + laneHi * 8 + 16];  // K {16..23}/{24..31}
            // load all four B fragments, then issue WMMAs back-to-back
            b0.h[0] = *(const v8bf*)(bPtr[0] + kk);
            b0.h[1] = *(const v8bf*)(bPtr[0] + kk + 16);
            b1.h[0] = *(const v8bf*)(bPtr[1] + kk);
            b1.h[1] = *(const v8bf*)(bPtr[1] + kk + 16);
            b2.h[0] = *(const v8bf*)(bPtr[2] + kk);
            b2.h[1] = *(const v8bf*)(bPtr[2] + kk + 16);
            b3.h[0] = *(const v8bf*)(bPtr[3] + kk);
            b3.h[1] = *(const v8bf*)(bPtr[3] + kk + 16);
            acc[0] = __builtin_amdgcn_wmma_f32_16x16x32_bf16(false, a.v, false, b0.v,
                                                             (short)0, acc[0], false, false);
            acc[1] = __builtin_amdgcn_wmma_f32_16x16x32_bf16(false, a.v, false, b1.v,
                                                             (short)0, acc[1], false, false);
            acc[2] = __builtin_amdgcn_wmma_f32_16x16x32_bf16(false, a.v, false, b2.v,
                                                             (short)0, acc[2], false, false);
            acc[3] = __builtin_amdgcn_wmma_f32_16x16x32_bf16(false, a.v, false, b3.v,
                                                             (short)0, acc[3], false, false);
        }
    }

    // ---- epilogue: bias + optional ReLU, coalesced NHWC stores
    const int colBase = nBase + waveN * 64 + l16;
    float bb[4];
#pragma unroll
    for (int s = 0; s < 4; ++s)
        bb[s] = (colBase + 16 * s < Cout) ? bias[colBase + 16 * s] : 0.f;
#pragma unroll
    for (int r = 0; r < 8; ++r) {
        const int m = mBase + waveM * 16 + laneHi * 8 + r;  // C/D layout: M = r + 8*half
        if (m >= Mtot) continue;
        const long rowOut = (long)m * Cout;
#pragma unroll
        for (int s = 0; s < 4; ++s) {
            const int col = colBase + 16 * s;
            if (col >= Cout) continue;
            float v = acc[s][r] + bb[s];
            if (relu) v = fmaxf(v, 0.f);
            if (Ybf) Ybf[rowOut + col] = (__bf16)v;
            else     Yf [rowOut + col] = v;
        }
    }
}

// ---------------------------------------------------------------------------
// Sigmoid-score histogram: 128 bins per image (approximate top-k selection)
// ---------------------------------------------------------------------------
__global__ void rn_score_hist(const __bf16* __restrict__ logits, int* __restrict__ hist,
                              int perImg, int total) {
    __shared__ int h[256];
    h[threadIdx.x] = 0;
    __syncthreads();
    for (int i = blockIdx.x * blockDim.x + threadIdx.x; i < total; i += gridDim.x * blockDim.x) {
        float x = (float)logits[i];
        float s = 1.f / (1.f + __expf(-x));
        int img = i / perImg;
        int b = (int)(s * 128.f);
        b = b < 0 ? 0 : (b > 127 ? 127 : b);
        atomicAdd(&h[img * 128 + b], 1);
    }
    __syncthreads();
    int v = h[threadIdx.x];
    if (v) atomicAdd(&hist[threadIdx.x], v);
}

__global__ void rn_cutoff(const int* __restrict__ hist, float* __restrict__ thr, int k) {
    int img = threadIdx.x;
    if (img >= RN_N) return;
    int acc = 0, b;
    for (b = 127; b >= 0; --b) {
        acc += hist[img * 128 + b];
        if (acc >= k) break;
    }
    if (b < 0) b = 0;
    thr[img] = fmaxf((float)b / 128.f, RN_SCORE_TH);
}

// ---------------------------------------------------------------------------
// Gather candidates above threshold (NHWC logits: [img][p][720])
// ---------------------------------------------------------------------------
__global__ void rn_gather(const __bf16* __restrict__ logits, const float* __restrict__ thr,
                          float* __restrict__ cScore, int* __restrict__ cAidx,
                          int* __restrict__ cCls, int* __restrict__ cCnt,
                          int HW, int lvl) {
    const int perImg = HW * RN_SCORE_CH;
    const int total = RN_N * perImg;
    for (int i = blockIdx.x * blockDim.x + threadIdx.x; i < total; i += gridDim.x * blockDim.x) {
        int img = i / perImg;
        int rem = i - img * perImg;
        int p = rem / RN_SCORE_CH;
        int ch = rem - p * RN_SCORE_CH;
        float x = (float)logits[i];
        float s = 1.f / (1.f + __expf(-x));
        if (s > thr[img]) {
            int slot = atomicAdd(&cCnt[img * 5 + lvl], 1);
            if (slot < RN_CAP) {
                int base = (img * 5 + lvl) * RN_CAP + slot;
                int a = ch / RN_K;
                int cls = ch - a * RN_K;
                cScore[base] = s;
                cAidx[base] = p * RN_A + a;  // (y*W+x)*A + a
                cCls[base] = cls;
            }
        }
    }
}

// ---------------------------------------------------------------------------
// Decode candidate boxes (anchors on the fly; NHWC deltas: [img][p][36])
// ---------------------------------------------------------------------------
__global__ void rn_decode(const float* __restrict__ deltas, float* __restrict__ cBox,
                          const int* __restrict__ cAidx, const int* __restrict__ cCnt,
                          int H, int Wd, int lvl, float stride) {
    int img = blockIdx.y;
    int cnt = cCnt[img * 5 + lvl];
    cnt = cnt < RN_CAP ? cnt : RN_CAP;
    int t = blockIdx.x * blockDim.x + threadIdx.x;
    if (t >= cnt) return;
    int base = (img * 5 + lvl) * RN_CAP + t;
    int aidx = cAidx[base];
    int p = aidx / RN_A;
    int a = aidx - p * RN_A;
    int y = p / Wd, x = p - y * Wd;
    int kz = a / 3, ari = a - kz * 3;
    float size = 4.f * stride * exp2f((float)kz / 3.f);
    float area = size * size;
    float ar = (ari == 0) ? 0.5f : ((ari == 1) ? 1.f : 2.f);
    float wa = sqrtf(area / ar);
    float ha = ar * wa;
    float cxa = ((float)x + 0.5f) * stride;
    float cya = ((float)y + 0.5f) * stride;
    const int HW = H * Wd;
    const float* dp = deltas + ((size_t)img * HW + p) * RN_PRED_CH + a * 4;
    float dx = dp[0];
    float dy = dp[1];
    float dw = fminf(dp[2], RN_SCALE_CLAMP);
    float dh = fminf(dp[3], RN_SCALE_CLAMP);
    float cx = dx * wa + cxa;
    float cy = dy * ha + cya;
    float wb = __expf(dw) * wa;
    float hb = __expf(dh) * ha;
    cBox[base * 4 + 0] = fminf(fmaxf(cx - 0.5f * wb, 0.f), RN_IMG_W);
    cBox[base * 4 + 1] = fminf(fmaxf(cy - 0.5f * hb, 0.f), RN_IMG_H);
    cBox[base * 4 + 2] = fminf(fmaxf(cx + 0.5f * wb, 0.f), RN_IMG_W);
    cBox[base * 4 + 3] = fminf(fmaxf(cy + 0.5f * hb, 0.f), RN_IMG_H);
}

// ---------------------------------------------------------------------------
// Class-aware greedy NMS, one block per image; writes dets + classes
// ---------------------------------------------------------------------------
__global__ __launch_bounds__(256)
void rn_nms(float* __restrict__ cScore, const float* __restrict__ cBox,
            const int* __restrict__ cCls, const int* __restrict__ cCnt,
            float* __restrict__ out) {
    const int img = blockIdx.x;
    const int t = threadIdx.x;
    const int TOT = 5 * RN_CAP;
    const int baseImg = img * TOT;
    __shared__ float bVal[256];
    __shared__ int   bIdx[256];
    __shared__ float sBox[4];
    __shared__ int   sCls;
    __shared__ int   sIdx;
    __shared__ float sSc;
    __shared__ int   vcnt[5];
    if (t < 5) {
        int c = cCnt[img * 5 + t];
        vcnt[t] = c < RN_CAP ? c : RN_CAP;
    }
    __syncthreads();

    for (int it = 0; it < RN_MAXDET; ++it) {
        float mv = -1e30f; int mi = -1;
        for (int j = t; j < TOT; j += 256) {
            int lvl = j / RN_CAP;
            int slot = j - lvl * RN_CAP;
            if (slot >= vcnt[lvl]) continue;
            float s = cScore[baseImg + j];
            if (s > mv) { mv = s; mi = j; }
        }
        bVal[t] = mv; bIdx[t] = mi;
        __syncthreads();
        for (int st = 128; st > 0; st >>= 1) {
            if (t < st && bVal[t + st] > bVal[t]) { bVal[t] = bVal[t + st]; bIdx[t] = bIdx[t + st]; }
            __syncthreads();
        }
        if (t == 0) {
            sSc = bVal[0]; sIdx = bIdx[0];
            if (sIdx >= 0 && sSc > 0.f) {
                sCls = cCls[baseImg + sIdx];
#pragma unroll
                for (int q = 0; q < 4; ++q) sBox[q] = cBox[(size_t)(baseImg + sIdx) * 4 + q];
            } else {
                sCls = -1;
            }
        }
        __syncthreads();
        // emit detection row (zeros / class -1 when exhausted)
        if (t < 5) {
            float v = 0.f;
            if (sCls >= 0) v = (t < 4) ? sBox[t] : sSc;
            out[((size_t)img * RN_MAXDET + it) * 5 + t] = v;
        }
        if (t == 8)
            out[RN_N * RN_MAXDET * 5 + img * RN_MAXDET + it] = (float)((sCls >= 0) ? sCls : -1);
        // suppress
        if (sCls >= 0) {
            float ax1 = sBox[0], ay1 = sBox[1], ax2 = sBox[2], ay2 = sBox[3];
            float aArea = (ax2 - ax1) * (ay2 - ay1);
            for (int j = t; j < TOT; j += 256) {
                int lvl = j / RN_CAP;
                int slot = j - lvl * RN_CAP;
                if (slot >= vcnt[lvl]) continue;
                if (j == sIdx) { cScore[baseImg + j] = -1.f; continue; }
                if (cCls[baseImg + j] != sCls) continue;
                float s = cScore[baseImg + j];
                if (s <= 0.f) continue;
                const float* bp = &cBox[(size_t)(baseImg + j) * 4];
                float ix1 = fmaxf(bp[0], ax1), iy1 = fmaxf(bp[1], ay1);
                float ix2 = fminf(bp[2], ax2), iy2 = fminf(bp[3], ay2);
                float inter = fmaxf(ix2 - ix1, 0.f) * fmaxf(iy2 - iy1, 0.f);
                float bArea = (bp[2] - bp[0]) * (bp[3] - bp[1]);
                float iou = inter / (aArea + bArea - inter + 1e-9f);
                if (iou >= RN_NMS_TH) cScore[baseImg + j] = -1.f;
            }
        }
        __syncthreads();
    }
}

// ---------------------------------------------------------------------------
// Host orchestration
// ---------------------------------------------------------------------------
extern "C" void kernel_launch(void* const* d_in, const int* in_sizes, int n_in,
                              void* d_out, int out_size, void* d_ws, size_t ws_size,
                              hipStream_t stream) {
    (void)in_sizes; (void)n_in; (void)out_size; (void)ws_size;

    static const int   LH[5] = {100, 50, 25, 13, 7};
    static const int   LW[5] = {152, 76, 38, 19, 10};
    static const float LS[5] = {8.f, 16.f, 32.f, 64.f, 128.f};

    const float* feat[5];
    for (int i = 0; i < 5; ++i) feat[i] = (const float*)d_in[i];
    const float* cls_w   = (const float*)d_in[5];
    const float* cls_b   = (const float*)d_in[6];
    const float* box_w   = (const float*)d_in[7];
    const float* box_b   = (const float*)d_in[8];
    const float* score_w = (const float*)d_in[9];
    const float* score_b = (const float*)d_in[10];
    const float* pred_w  = (const float*)d_in[11];
    const float* pred_b  = (const float*)d_in[12];
    float* out = (float*)d_out;

    // padded Cout (to 128-wide N tiles) for guard-free B fragment loads
    const int SCORE_PAD = 768;   // 720 -> 768
    const int PRED_PAD  = 128;   // 36  -> 128

    // ---- workspace carve-up (256B aligned)
    size_t off = 0;
    auto alloc = [&](size_t bytes) -> void* {
        off = (off + 255) & ~(size_t)255;
        void* p = (char*)d_ws + off;
        off += bytes;
        return p;
    };
    const size_t maxHW = (size_t)LH[0] * LW[0];                       // 15200
    __bf16* wtCls   = (__bf16*)alloc((size_t)4 * RN_C * RN_KTOT * 2);
    __bf16* wtBox   = (__bf16*)alloc((size_t)4 * RN_C * RN_KTOT * 2);
    __bf16* wtScore = (__bf16*)alloc((size_t)SCORE_PAD * RN_KTOT * 2);
    __bf16* wtPred  = (__bf16*)alloc((size_t)PRED_PAD * RN_KTOT * 2);
    __bf16* featbf  = (__bf16*)alloc((size_t)RN_N * RN_C * maxHW * 2);
    __bf16* actA    = (__bf16*)alloc((size_t)RN_N * RN_C * maxHW * 2);
    __bf16* actB    = (__bf16*)alloc((size_t)RN_N * RN_C * maxHW * 2);
    __bf16* scoreL  = (__bf16*)alloc((size_t)RN_N * RN_SCORE_CH * maxHW * 2);
    float*  deltas  = (float*) alloc((size_t)RN_N * RN_PRED_CH * maxHW * 4);
    int*    hist    = (int*)   alloc(256 * sizeof(int));
    float*  thr     = (float*) alloc(RN_N * sizeof(float));
    int*    cCnt    = (int*)   alloc(RN_N * 5 * sizeof(int));
    float*  cScore  = (float*) alloc((size_t)RN_N * 5 * RN_CAP * 4);
    int*    cAidx   = (int*)   alloc((size_t)RN_N * 5 * RN_CAP * 4);
    int*    cCls    = (int*)   alloc((size_t)RN_N * 5 * RN_CAP * 4);
    float*  cBox    = (float*) alloc((size_t)RN_N * 5 * RN_CAP * 16);

    // ---- repack weights to bf16 column-major tap-major layout (zero padded)
    {
        int tot = RN_C * RN_KTOT;
        int blocks = (tot + 255) / 256;
        for (int i = 0; i < 4; ++i) {
            rn_wtransform<<<blocks, 256, 0, stream>>>(
                cls_w + (size_t)i * RN_C * RN_C * 9, wtCls + (size_t)i * RN_C * RN_KTOT,
                RN_C, RN_C);
            rn_wtransform<<<blocks, 256, 0, stream>>>(
                box_w + (size_t)i * RN_C * RN_C * 9, wtBox + (size_t)i * RN_C * RN_KTOT,
                RN_C, RN_C);
        }
        int totS = SCORE_PAD * RN_KTOT;
        rn_wtransform<<<(totS + 255) / 256, 256, 0, stream>>>(score_w, wtScore,
                                                              RN_SCORE_CH, SCORE_PAD);
        int totP = PRED_PAD * RN_KTOT;
        rn_wtransform<<<(totP + 255) / 256, 256, 0, stream>>>(pred_w, wtPred,
                                                              RN_PRED_CH, PRED_PAD);
    }

    hipMemsetAsync(cCnt, 0, RN_N * 5 * sizeof(int), stream);

    for (int lvl = 0; lvl < 5; ++lvl) {
        const int H = LH[lvl], Wd = LW[lvl];
        const int HW = H * Wd;
        const int Mtot = RN_N * HW;
        const int mTiles = (Mtot + 63) / 64;

        // feature NCHW f32 -> NHWC bf16
        {
            int n = RN_N * RN_C * HW;
            int blocks = (n + 255) / 256;
            if (blocks > 2048) blocks = 2048;
            rn_f2bf_nhwc<<<blocks, 256, 0, stream>>>(feat[lvl], featbf, HW);
        }

        dim3 g256(mTiles, (RN_C + 127) / 128);
        // ---- cls subnet: feat -> A -> B -> A -> B
        rn_conv3x3_wmma<<<g256, 256, 0, stream>>>(featbf, wtCls + 0 * (size_t)RN_C * RN_KTOT,
            cls_b + 0 * RN_C, actA, nullptr, RN_N, H, Wd, RN_C, 1);
        rn_conv3x3_wmma<<<g256, 256, 0, stream>>>(actA, wtCls + 1 * (size_t)RN_C * RN_KTOT,
            cls_b + 1 * RN_C, actB, nullptr, RN_N, H, Wd, RN_C, 1);
        rn_conv3x3_wmma<<<g256, 256, 0, stream>>>(actB, wtCls + 2 * (size_t)RN_C * RN_KTOT,
            cls_b + 2 * RN_C, actA, nullptr, RN_N, H, Wd, RN_C, 1);
        rn_conv3x3_wmma<<<g256, 256, 0, stream>>>(actA, wtCls + 3 * (size_t)RN_C * RN_KTOT,
            cls_b + 3 * RN_C, actB, nullptr, RN_N, H, Wd, RN_C, 1);
        // score head (720 ch, bf16 logits, no relu)
        dim3 gScore(mTiles, SCORE_PAD / 128);
        rn_conv3x3_wmma<<<gScore, 256, 0, stream>>>(actB, wtScore, score_b,
            scoreL, nullptr, RN_N, H, Wd, RN_SCORE_CH, 0);

        // ---- box subnet: feat -> A -> B -> A -> B
        rn_conv3x3_wmma<<<g256, 256, 0, stream>>>(featbf, wtBox + 0 * (size_t)RN_C * RN_KTOT,
            box_b + 0 * RN_C, actA, nullptr, RN_N, H, Wd, RN_C, 1);
        rn_conv3x3_wmma<<<g256, 256, 0, stream>>>(actA, wtBox + 1 * (size_t)RN_C * RN_KTOT,
            box_b + 1 * RN_C, actB, nullptr, RN_N, H, Wd, RN_C, 1);
        rn_conv3x3_wmma<<<g256, 256, 0, stream>>>(actB, wtBox + 2 * (size_t)RN_C * RN_KTOT,
            box_b + 2 * RN_C, actA, nullptr, RN_N, H, Wd, RN_C, 1);
        rn_conv3x3_wmma<<<g256, 256, 0, stream>>>(actA, wtBox + 3 * (size_t)RN_C * RN_KTOT,
            box_b + 3 * RN_C, actB, nullptr, RN_N, H, Wd, RN_C, 1);
        // pred head (36 ch, f32 deltas, no relu)
        dim3 gPred(mTiles, PRED_PAD / 128);
        rn_conv3x3_wmma<<<gPred, 256, 0, stream>>>(actB, wtPred, pred_b,
            nullptr, deltas, RN_N, H, Wd, RN_PRED_CH, 0);

        // ---- approximate top-k selection
        hipMemsetAsync(hist, 0, 256 * sizeof(int), stream);
        {
            int total = RN_N * RN_SCORE_CH * HW;
            int blocks = (total + 255) / 256;
            if (blocks > 1024) blocks = 1024;
            rn_score_hist<<<blocks, 256, 0, stream>>>(scoreL, hist, RN_SCORE_CH * HW, total);
            rn_cutoff<<<1, 32, 0, stream>>>(hist, thr, RN_TOPK);
            rn_gather<<<blocks, 256, 0, stream>>>(scoreL, thr, cScore, cAidx, cCls, cCnt,
                                                  HW, lvl);
        }
        // ---- decode boxes for this level
        dim3 gDec((RN_CAP + 255) / 256, RN_N);
        rn_decode<<<gDec, 256, 0, stream>>>(deltas, cBox, cAidx, cCnt, H, Wd, lvl, LS[lvl]);
    }

    // ---- NMS + output
    rn_nms<<<RN_N, 256, 0, stream>>>(cScore, cBox, cCls, cCnt, out);
}
